// AMRSRN_36945308680687
// MI455X (gfx1250) — compile-verified, matrix operand order checked by reading, not run on hardware
//
#include <hip/hip_runtime.h>

typedef __attribute__((ext_vector_type(16))) _Float16 v16h;
typedef __attribute__((ext_vector_type(8)))  float    v8f;

#define INV_148   (1.0f / 1.48f)
#define GRIDS     32
#define GRID_VOL  262144      // 64*64*64
#define PTS_PER_BLOCK 128     // 8 waves * 16 points

// Build the two A fragments (K=0..31 and K=32..63) for a 16x64 f16 activation
// tile stored row-major in LDS (hb[m*64 + k]), following the CDNA5 16-bit
// A-matrix 16x32 layout: lane = {m = lane&15, h = lane>>4},
// VGPR v (element pair e=2v): K = (v<4 ? 0 : 16) + 8*h + 2*(v&3) + {0,1}.
__device__ __forceinline__ void load_a_frags(const _Float16* hb, int m, int halfId,
                                             v16h& a0, v16h& a1) {
#pragma unroll
  for (int e = 0; e < 16; e += 2) {
    int v = e >> 1;
    int k = ((v < 4) ? 0 : 16) + halfId * 8 + 2 * (v & 3);
    a0[e]     = hb[m * 64 + k];
    a0[e + 1] = hb[m * 64 + k + 1];
    a1[e]     = hb[m * 64 + 32 + k];
    a1[e + 1] = hb[m * 64 + 32 + k + 1];
  }
}

// One 64->64 layer for a wave's 16-point tile: D = relu(A(16x64) * W^T + b).
// W is stored row-major f16 in LDS as W[n*64 + k] (out[n] = sum_k a[k]*W[n][k]),
// so B[k][n] = W[n][k]; B fragment uses the same K packing as A with lane -> n.
__device__ __forceinline__ void mlp_layer64(_Float16* hb, const _Float16* sW,
                                            const float* sBias, int m, int halfId) {
  v16h a0, a1;
  load_a_frags(hb, m, halfId, a0, a1);

  v8f d[4];
#pragma unroll
  for (int t = 0; t < 4; ++t) {
    int n = t * 16 + m;          // output column this lane owns
    v16h bk0, bk1;
#pragma unroll
    for (int e = 0; e < 16; e += 2) {
      int v = e >> 1;
      int k = ((v < 4) ? 0 : 16) + halfId * 8 + 2 * (v & 3);
      bk0[e]     = sW[n * 64 + k];
      bk0[e + 1] = sW[n * 64 + k + 1];
      bk1[e]     = sW[n * 64 + 32 + k];
      bk1[e + 1] = sW[n * 64 + 32 + k + 1];
    }
    v8f c = {};
    c = __builtin_amdgcn_wmma_f32_16x16x32_f16(false, a0, false, bk0, (short)0, c, false, false);
    c = __builtin_amdgcn_wmma_f32_16x16x32_f16(false, a1, false, bk1, (short)0, c, false, false);
    float bias = sBias[n];
#pragma unroll
    for (int r = 0; r < 8; ++r)
      d[t][r] = fmaxf(c[r] + bias, 0.0f);
  }
  __syncthreads();   // everyone done reading hb before overwrite
  // C/D layout: VGPR r, lanes 0-15 -> M=r, lanes 16-31 -> M=r+8; N = lane&15 (+16t)
#pragma unroll
  for (int t = 0; t < 4; ++t)
#pragma unroll
    for (int r = 0; r < 8; ++r)
      hb[(r + 8 * halfId) * 64 + t * 16 + m] = (_Float16)d[t][r];
  __syncthreads();
}

__global__ __launch_bounds__(256) void amrsrn_fused_kernel(
    const float* __restrict__ x,           // (N,3)
    const float* __restrict__ grid_scales, // (32,3)
    const float* __restrict__ grid_trans,  // (32,3)
    const float* __restrict__ grids,       // (32,2,64,64,64)
    const float* __restrict__ W0, const float* __restrict__ b0,   // (64,64),(64)
    const float* __restrict__ W1, const float* __restrict__ b1,   // (64,64),(64)
    const float* __restrict__ W2, const float* __restrict__ b2,   // (1,64),(1)
    float* __restrict__ out, int N)
{
  __shared__ _Float16 sW0h[64 * 64];
  __shared__ _Float16 sW1h[64 * 64];
  __shared__ float    sB0[64], sB1[64], sW2v[64];
  __shared__ float    sScale[GRIDS * 3], sTrans[GRIDS * 3];
  __shared__ _Float16 sH[8][16 * 64];    // per-wave activation tile (16 pts x 64)
  __shared__ float    sB2v;

  const int tid = threadIdx.x;

  // Stage weights once per block: f32 -> f16 conversion into LDS.
  for (int i = tid; i < 64 * 64; i += 256) {
    sW0h[i] = (_Float16)W0[i];
    sW1h[i] = (_Float16)W1[i];
  }
  if (tid < 64) { sB0[tid] = b0[tid]; sB1[tid] = b1[tid]; sW2v[tid] = W2[tid]; }
  if (tid < GRIDS * 3) { sScale[tid] = grid_scales[tid]; sTrans[tid] = grid_trans[tid]; }
  if (tid == 0) sB2v = b2[0];
  __syncthreads();

  const int wave   = tid >> 5;
  const int lane   = tid & 31;
  const int m      = lane & 15;   // point within the wave tile
  const int halfId = lane >> 4;   // which half of the 64 features / K-range
  const int P      = blockIdx.x * PTS_PER_BLOCK + wave * 16 + m;
  const int Pc     = (P < N) ? P : (N - 1);

  const float px = x[Pc * 3 + 0];
  const float py = x[Pc * 3 + 1];
  const float pz = x[Pc * 3 + 2];

  _Float16* hb = sH[wave];

  // ---- Phase 1: trilinear sampling. 2 lanes per point; each lane does 16 grids.
  for (int gi = 0; gi < 16; ++gi) {
    const int g = halfId * 16 + gi;
    float fx = (px * sScale[g * 3 + 0] + sTrans[g * 3 + 0]) * INV_148;
    float fy = (py * sScale[g * 3 + 1] + sTrans[g * 3 + 1]) * INV_148;
    float fz = (pz * sScale[g * 3 + 2] + sTrans[g * 3 + 2]) * INV_148;
    fx = (fx + 1.0f) * 0.5f * 63.0f;
    fy = (fy + 1.0f) * 0.5f * 63.0f;
    fz = (fz + 1.0f) * 0.5f * 63.0f;
    const float x0f = floorf(fx), y0f = floorf(fy), z0f = floorf(fz);
    const int   ix = (int)x0f, iy = (int)y0f, iz = (int)z0f;
    const float wx = fx - x0f, wy = fy - y0f, wz = fz - z0f;

    const float* g0 = grids + (size_t)(g * 2) * GRID_VOL;
    float acc0 = 0.0f, acc1 = 0.0f;
#pragma unroll
    for (int dz = 0; dz < 2; ++dz) {
      const int zi = iz + dz;
      const bool vz = (zi >= 0) & (zi < 64);
      const int  zc = min(max(zi, 0), 63);
      const float wzv = dz ? wz : 1.0f - wz;
#pragma unroll
      for (int dy = 0; dy < 2; ++dy) {
        const int yi = iy + dy;
        const bool vy = (yi >= 0) & (yi < 64);
        const int  yc = min(max(yi, 0), 63);
        const float wyv = dy ? wy : 1.0f - wy;
        const int rowOff = (zc * 64 + yc) * 64;
#pragma unroll
        for (int dx = 0; dx < 2; ++dx) {
          const int xi = ix + dx;
          const bool vx = (xi >= 0) & (xi < 64);
          const int  xc = min(max(xi, 0), 63);
          const float wxv = dx ? wx : 1.0f - wx;
          const float w = wxv * wyv * wzv * (float)(vx & vy & vz);
          const int off = rowOff + xc;
          acc0 = fmaf(g0[off], w, acc0);
          acc1 = fmaf(g0[GRID_VOL + off], w, acc1);
        }
      }
    }
    hb[m * 64 + g * 2 + 0] = (_Float16)acc0;
    hb[m * 64 + g * 2 + 1] = (_Float16)acc1;
  }
  __syncthreads();

  // ---- Phase 2/3: two fused 64->64 WMMA layers with ReLU.
  mlp_layer64(hb, sW0h, sB0, m, halfId);
  mlp_layer64(hb, sW1h, sB1, m, halfId);

  // ---- Phase 4: final 64->1 projection (lanes 0-15, one point each).
  if (halfId == 0 && P < N) {
    float s = sB2v;
#pragma unroll
    for (int k = 0; k < 64; ++k)
      s = fmaf((float)hb[m * 64 + k], sW2v[k], s);
    out[P] = s;
  }
}

extern "C" void kernel_launch(void* const* d_in, const int* in_sizes, int n_in,
                              void* d_out, int out_size, void* d_ws, size_t ws_size,
                              hipStream_t stream) {
  const float* x      = (const float*)d_in[0];
  const float* scales = (const float*)d_in[1];
  const float* trans  = (const float*)d_in[2];
  const float* grids  = (const float*)d_in[3];
  const float* W0     = (const float*)d_in[4];
  const float* b0     = (const float*)d_in[5];
  const float* W1     = (const float*)d_in[6];
  const float* b1     = (const float*)d_in[7];
  const float* W2     = (const float*)d_in[8];
  const float* b2     = (const float*)d_in[9];
  float* out = (float*)d_out;

  const int N = in_sizes[0] / 3;
  const int nBlocks = (N + PTS_PER_BLOCK - 1) / PTS_PER_BLOCK;
  amrsrn_fused_kernel<<<nBlocks, 256, 0, stream>>>(
      x, scales, trans, grids, W0, b0, W1, b1, W2, b2, out, N);
}
